// PitchSequenceEncoder_72069551226904
// MI455X (gfx1250) — matrix operand order repaired and checked by reading, not compiled, not run on hardware
//
#include <hip/hip_runtime.h>
#include <math.h>

typedef __attribute__((ext_vector_type(16))) __bf16 v16bf;
typedef __attribute__((ext_vector_type(8)))  float  v8f;
typedef unsigned int v4u __attribute__((ext_vector_type(4)));
typedef int          v8i __attribute__((ext_vector_type(8)));
typedef int          v4i __attribute__((ext_vector_type(4)));

static constexpr int B_ROWS = 32768;
static constexpr int IN_DIM = 288;     // 32 numeric + 64 + 128 + 64 one-hot
static constexpr int H1D    = 4096;
static constexpr int H2D    = 2048;
static constexpr int H3D    = 1024;
static constexpr int NCLS   = 19;

#if defined(__has_builtin)
#if __has_builtin(__builtin_amdgcn_tensor_load_to_lds) && \
    __has_builtin(__builtin_amdgcn_s_wait_tensorcnt)
#define PSE_USE_TDM 1
#endif
#endif
#ifndef PSE_USE_TDM
#define PSE_USE_TDM 0
#endif

// ---------------------------------------------------------------- utilities
__global__ __launch_bounds__(256) void cvt_f32_bf16(const float* __restrict__ src,
                                                    __bf16* __restrict__ dst, int n) {
  int i = blockIdx.x * 256 + threadIdx.x;
  if (i < n) dst[i] = (__bf16)src[i];
}

__global__ __launch_bounds__(256) void zero_f32(float* __restrict__ p, int n) {
  int i = blockIdx.x * 256 + threadIdx.x;
  if (i < n) p[i] = 0.0f;
}

// one-hot expansion: X[B, 288] bf16
__global__ __launch_bounds__(256) void encode_x(const float* __restrict__ numeric,
                                                const int* __restrict__ c0,
                                                const int* __restrict__ c1,
                                                const int* __restrict__ c2,
                                                __bf16* __restrict__ X) {
  int i = blockIdx.x * 256 + threadIdx.x;
  if (i >= B_ROWS * IN_DIM) return;
  int row = i / IN_DIM;
  int col = i - row * IN_DIM;
  float v;
  if (col < 32)       v = numeric[row * 32 + col];
  else if (col < 96)  v = (c0[row] == col - 32)  ? 1.0f : 0.0f;
  else if (col < 224) v = (c1[row] == col - 96)  ? 1.0f : 0.0f;
  else                v = (c2[row] == col - 224) ? 1.0f : 0.0f;
  X[i] = (__bf16)v;
}

#if PSE_USE_TDM
// TDM: DMA a 64-row x KC-col bf16 tile (row stride K elems) from global into
// LDS at byte address ldsByte, inserting 16B padding after each row so the
// LDS layout has stride (KC + 8) bf16 elements (bank-conflict padding).
__device__ __forceinline__ void tdm_load_A(const __bf16* gsrc, unsigned int ldsByte,
                                           int KC, int K) {
  unsigned long long ga = (unsigned long long)(uintptr_t)gsrc;
  // pad_interval code: pad after KC elems = KC/2 dwords = 2^(log2(KC)-1)
  // table: code c -> 2^(c+1) dwords  =>  c = log2(KC) - 2
  int pic = 31 - __builtin_clz((unsigned)KC) - 2;
  v4u g0;
  g0[0] = 1u;                                        // count=1 (valid), user mode
  g0[1] = ldsByte;                                   // lds_addr (bytes)
  g0[2] = (unsigned int)ga;                          // global_addr[31:0]
  g0[3] = (unsigned int)((ga >> 32) & 0x01FFFFFFu)   // global_addr[56:32]
        | (2u << 30);                                // type = 2 ("image")
  v8i g1;
  g1[0] = (int)((1u << 16)                           // data_size = 2 bytes
              | (1u << 20)                           // pad_enable
              | ((unsigned int)pic << 22)            // pad_interval
              | (3u << 25));                         // pad_amount = 4 dwords (16B)
  g1[1] = (int)(((unsigned int)KC & 0xFFFFu) << 16); // tensor_dim0[15:0] (bits 63:48)
  g1[2] = (int)(64u << 16);                          // tensor_dim0 hi=0 | tensor_dim1 lo=64
  g1[3] = (int)(((unsigned int)KC & 0xFFFFu) << 16); // tensor_dim1 hi=0 | tile_dim0=KC
  g1[4] = 64;                                        // tile_dim1=64, tile_dim2=0
  g1[5] = K;                                         // tensor_dim0_stride[31:0] = K elems
  g1[6] = 0;                                         // stride hi | dim1_stride lo
  g1[7] = 0;
  v4i gz = {0, 0, 0, 0};
#if __clang_major__ >= 23
  v8i gz8 = {0, 0, 0, 0, 0, 0, 0, 0};
  __builtin_amdgcn_tensor_load_to_lds(g0, g1, gz, gz, gz8, 0);
#else
  __builtin_amdgcn_tensor_load_to_lds(g0, g1, gz, gz, 0);
#endif
}
#endif  // PSE_USE_TDM

// ------------------------------------------------- WMMA GEMM + GELU + stats
// Block: 64 rows x 256 cols, 256 threads (8 waves). Wave w: cols [w*32, w*32+32).
// Per wave: 4 row-groups x 2 col-tiles of v_wmma_f32_16x16x32_bf16.
// A tiles staged in LDS in KC=128 chunks; with TDM, chunk c+1 is DMA'd
// (TENSORcnt) while chunk c is computed (double buffer).
__global__ __launch_bounds__(256)
void gemm_bf16_gelu_stats(const __bf16* __restrict__ X,   // [B, K] bf16 row-major
                          const __bf16* __restrict__ W,   // [K, N] bf16 row-major
                          const float*  __restrict__ bias,
                          __bf16* __restrict__ Y,         // [B, N] bf16 (post-GELU, pre-LN)
                          float* __restrict__ rowsum,
                          float* __restrict__ rowssq,
                          int K, int N) {
  __shared__ __align__(16) __bf16 sA[2][64 * 136];  // 2 x (64 rows x (128+8)) bf16
  __shared__ float sStat[128];                      // sum[0..63], ssq[64..127]

  const int tid  = threadIdx.x;
  const int lane = tid & 31;
  const int wave = tid >> 5;
  const int m0   = blockIdx.x * 64;
  const int ncol0 = blockIdx.y * 256 + wave * 32;
  const int ml = lane & 15;       // A: row-in-group / B: N column ; C: column lane
  const int h  = lane >> 4;       // half selector per ISA layouts

  if (tid < 128) sStat[tid] = 0.0f;

  v8f acc[4][2];
#pragma unroll
  for (int g = 0; g < 4; ++g)
#pragma unroll
    for (int t = 0; t < 2; ++t)
#pragma unroll
      for (int r = 0; r < 8; ++r) acc[g][t][r] = 0.0f;

  const int nch = (K + 127) >> 7;

#if PSE_USE_TDM
  if (wave == 0) {
    int KC0 = K < 128 ? K : 128;
    tdm_load_A(X + (size_t)m0 * K, (unsigned int)(uintptr_t)&sA[0][0], KC0, K);
  }
#endif

  for (int c = 0; c < nch; ++c) {
    const int kc  = c << 7;
    const int KC  = (K - kc) < 128 ? (K - kc) : 128;   // 128 or 32 (K=288 tail)
    const int SAS = KC + 8;                            // padded LDS stride (elements)
    __bf16* buf = sA[c & 1];

#if PSE_USE_TDM
    if (wave == 0) __builtin_amdgcn_s_wait_tensorcnt(0);  // chunk c landed
    __syncthreads();                                      // visible to all waves;
                                                          // also: compute c-1 done
    if (wave == 0 && c + 1 < nch) {                       // prefetch next chunk
      int kcn = kc + 128;
      int KCn = (K - kcn) < 128 ? (K - kcn) : 128;
      tdm_load_A(X + (size_t)m0 * K + kcn,
                 (unsigned int)(uintptr_t)&sA[(c + 1) & 1][0], KCn, K);
    }
#else
    __syncthreads();                                   // previous chunk consumed
    const int cpr = KC >> 3;                           // uint4 chunks per row
    for (int cc = tid; cc < 64 * cpr; cc += 256) {
      int row = cc / cpr;
      int kk  = (cc - row * cpr) << 3;
      *(uint4*)(buf + row * SAS + kk) =
          *(const uint4*)(X + (size_t)(m0 + row) * K + kc + kk);
    }
    __syncthreads();
#endif

    const __bf16* apb = buf + ml * SAS + 8 * h;
    for (int k2 = 0; k2 < KC; k2 += 32) {
      union { v16bf v; uint4 q[2]; } a[4], b[2];
      // A fragments (16-bit A 16x32 layout): lane=(m,h): K = 8h..8h+7, 16+8h..+7
#pragma unroll
      for (int g = 0; g < 4; ++g) {
        const __bf16* ap = apb + g * 16 * SAS + k2;
        a[g].q[0] = *(const uint4*)(ap);
        a[g].q[1] = *(const uint4*)(ap + 16);
      }
      // B fragments: lane = N column ml, K = 8h..8h+7, 16+8h..+7 (matches A)
      const __bf16* wcol = W + (size_t)(kc + k2 + 8 * h) * N + ncol0 + ml;
#pragma unroll
      for (int t = 0; t < 2; ++t) {
#pragma unroll
        for (int s = 0; s < 16; ++s)
          b[t].v[s] = wcol[(size_t)((s & 7) + ((s >> 3) << 4)) * N + t * 16];
      }
#pragma unroll
      for (int t = 0; t < 2; ++t)
#pragma unroll
        for (int g = 0; g < 4; ++g)
          acc[g][t] = __builtin_amdgcn_wmma_f32_16x16x32_bf16(
              false, a[g].v, false, b[t].v, (short)0, acc[g][t], false, false);
    }
  }

  // epilogue: bias + exact GELU, bf16 store, per-row stat partials (per group
  // to bound live registers)
#pragma unroll
  for (int g = 0; g < 4; ++g) {
    float psum[8], pssq[8];
#pragma unroll
    for (int r = 0; r < 8; ++r) { psum[r] = 0.0f; pssq[r] = 0.0f; }
#pragma unroll
    for (int t = 0; t < 2; ++t) {
      const int col = ncol0 + t * 16 + ml;   // C layout: lane column fixed
      const float bc = bias[col];
#pragma unroll
      for (int r = 0; r < 8; ++r) {          // C layout: VGPR r -> row r + 8h
        float y  = acc[g][t][r] + bc;
        float gl = 0.5f * y * (1.0f + erff(y * 0.70710678118654752f));
        psum[r] += gl;
        pssq[r] += gl * gl;
        Y[(size_t)(m0 + g * 16 + r + 8 * h) * N + col] = (__bf16)gl;
      }
    }
#pragma unroll
    for (int r = 0; r < 8; ++r) {
      int row = g * 16 + r + 8 * h;
      atomicAdd(&sStat[row],      psum[r]);
      atomicAdd(&sStat[64 + row], pssq[r]);
    }
  }
  __syncthreads();
  if (tid < 64) {
    atomicAdd(&rowsum[m0 + tid], sStat[tid]);
    atomicAdd(&rowssq[m0 + tid], sStat[64 + tid]);
  }
}

// ------------------------------------------------------------ LayerNorm apply
__global__ __launch_bounds__(256)
void ln_apply(__bf16* __restrict__ Y, const float* __restrict__ rowsum,
              const float* __restrict__ rowssq, const float* __restrict__ gamma,
              const float* __restrict__ beta, int N, int total8) {
  int i = blockIdx.x * 256 + threadIdx.x;
  if (i >= total8) return;
  int e0  = i << 3;
  int row = e0 / N;
  int col = e0 - row * N;
  float inv  = 1.0f / (float)N;
  float mean = rowsum[row] * inv;
  float var  = rowssq[row] * inv - mean * mean;
  float rs   = rsqrtf(var + 1e-5f);
  uint4 q = *(uint4*)(Y + (size_t)e0);
  __bf16* p = (__bf16*)&q;
#pragma unroll
  for (int j = 0; j < 8; ++j) {
    float v = (float)p[j];
    p[j] = (__bf16)((v - mean) * rs * gamma[col + j] + beta[col + j]);
  }
  *(uint4*)(Y + (size_t)e0) = q;
}

// ----------------------------------------- head: logits + mask + softmax/argmax
__global__ __launch_bounds__(256)
void head_softmax(const __bf16* __restrict__ Hin,   // [B, 1024] bf16
                  const float* __restrict__ W4,     // [1024, 19] f32
                  const float* __restrict__ b4,
                  const float* __restrict__ amask,  // [1000, 19]
                  const int*   __restrict__ pid,
                  float* __restrict__ out) {
  __shared__ __bf16 sW[H3D * NCLS];                 // ~38 KB
  const int tid = threadIdx.x, lane = tid & 31, wave = tid >> 5;
  for (int i = tid; i < H3D * NCLS; i += 256) sW[i] = (__bf16)W4[i];
  __syncthreads();

  const int rowBase = (blockIdx.x * 8 + wave) * 4;  // 4 rows per wave
  float acc0 = 0.f, acc1 = 0.f, acc2 = 0.f, acc3 = 0.f;
  for (int kk = 0; kk < H3D; kk += 32) {
    float v0 = (float)Hin[(size_t)(rowBase + 0) * H3D + kk + lane];
    float v1 = (float)Hin[(size_t)(rowBase + 1) * H3D + kk + lane];
    float v2 = (float)Hin[(size_t)(rowBase + 2) * H3D + kk + lane];
    float v3 = (float)Hin[(size_t)(rowBase + 3) * H3D + kk + lane];
#pragma unroll
    for (int j = 0; j < 32; ++j) {
      float w = (lane < NCLS) ? (float)sW[(kk + j) * NCLS + lane] : 0.0f;
      acc0 += __shfl(v0, j, 32) * w;
      acc1 += __shfl(v1, j, 32) * w;
      acc2 += __shfl(v2, j, 32) * w;
      acc3 += __shfl(v3, j, 32) * w;
    }
  }

  float accs[4] = {acc0, acc1, acc2, acc3};
#pragma unroll
  for (int rr = 0; rr < 4; ++rr) {
    int row = rowBase + rr;
    float logit = (lane < NCLS)
                      ? accs[rr] + b4[lane] + amask[pid[row] * NCLS + lane]
                      : -3.0e38f;
    float mx = logit;
    for (int off = 16; off > 0; off >>= 1) mx = fmaxf(mx, __shfl_xor(mx, off, 32));
    float e = (lane < NCLS) ? expf(logit - mx) : 0.0f;
    float s = e;
    for (int off = 16; off > 0; off >>= 1) s += __shfl_xor(s, off, 32);
    float prob = e / s;
    // argmax with first-index tie-break
    float av = (lane < NCLS) ? prob : -1.0f;
    int   ai = lane;
    for (int off = 16; off > 0; off >>= 1) {
      float ov = __shfl_xor(av, off, 32);
      int   oi = __shfl_xor(ai, off, 32);
      if (ov > av || (ov == av && oi < ai)) { av = ov; ai = oi; }
    }
    if (lane < NCLS) {
      out[(size_t)row * NCLS + lane] = logit;                               // masked logits
      out[(size_t)B_ROWS * NCLS + (size_t)row * NCLS + lane] = prob;        // probs
    }
    if (lane == 0) out[(size_t)2 * B_ROWS * NCLS + row] = (float)ai;        // pred
  }
}

// ------------------------------------------------------------------- launch
extern "C" void kernel_launch(void* const* d_in, const int* in_sizes, int n_in,
                              void* d_out, int out_size, void* d_ws, size_t ws_size,
                              hipStream_t stream) {
  (void)in_sizes; (void)n_in; (void)out_size; (void)ws_size;
  const float* numeric = (const float*)d_in[0];
  const int*   cat0    = (const int*)d_in[1];
  const int*   cat1    = (const int*)d_in[2];
  const int*   cat2    = (const int*)d_in[3];
  const int*   pid     = (const int*)d_in[4];
  const float* amask   = (const float*)d_in[5];
  const float* W1 = (const float*)d_in[6];  const float* b1  = (const float*)d_in[7];
  const float* g1 = (const float*)d_in[8];  const float* be1 = (const float*)d_in[9];
  const float* W2 = (const float*)d_in[10]; const float* b2  = (const float*)d_in[11];
  const float* g2 = (const float*)d_in[12]; const float* be2 = (const float*)d_in[13];
  const float* W3 = (const float*)d_in[14]; const float* b3  = (const float*)d_in[15];
  const float* g3 = (const float*)d_in[16]; const float* be3 = (const float*)d_in[17];
  const float* W4 = (const float*)d_in[18]; const float* b4  = (const float*)d_in[19];

  char* ws = (char*)d_ws;
  size_t off = 0;
  auto carve = [&](size_t bytes) -> char* {
    char* p = ws + off;
    off = (off + bytes + 255) & ~(size_t)255;
    return p;
  };
  __bf16* Xb   = (__bf16*)carve((size_t)B_ROWS * IN_DIM * 2);
  __bf16* W1b  = (__bf16*)carve((size_t)IN_DIM * H1D * 2);
  __bf16* W2b  = (__bf16*)carve((size_t)H1D * H2D * 2);
  __bf16* W3b  = (__bf16*)carve((size_t)H2D * H3D * 2);
  __bf16* Hb   = (__bf16*)carve((size_t)B_ROWS * H1D * 2);
  __bf16* Eb   = (__bf16*)carve((size_t)B_ROWS * H2D * 2);
  __bf16* H3b  = (__bf16*)carve((size_t)B_ROWS * H3D * 2);
  float*  rsum = (float*)carve((size_t)B_ROWS * 4);
  float*  rssq = (float*)carve((size_t)B_ROWS * 4);

  auto cdiv = [](long a, long b) { return (int)((a + b - 1) / b); };

  // weight conversion + input encode
  cvt_f32_bf16<<<cdiv((long)IN_DIM * H1D, 256), 256, 0, stream>>>(W1, W1b, IN_DIM * H1D);
  cvt_f32_bf16<<<cdiv((long)H1D * H2D, 256), 256, 0, stream>>>(W2, W2b, H1D * H2D);
  cvt_f32_bf16<<<cdiv((long)H2D * H3D, 256), 256, 0, stream>>>(W3, W3b, H2D * H3D);
  encode_x<<<cdiv((long)B_ROWS * IN_DIM, 256), 256, 0, stream>>>(numeric, cat0, cat1, cat2, Xb);

  // layer 1: [B,288] x [288,4096]
  zero_f32<<<cdiv(B_ROWS, 256), 256, 0, stream>>>(rsum, B_ROWS);
  zero_f32<<<cdiv(B_ROWS, 256), 256, 0, stream>>>(rssq, B_ROWS);
  gemm_bf16_gelu_stats<<<dim3(B_ROWS / 64, H1D / 256), 256, 0, stream>>>(
      Xb, W1b, b1, Hb, rsum, rssq, IN_DIM, H1D);
  ln_apply<<<cdiv((long)B_ROWS * H1D / 8, 256), 256, 0, stream>>>(
      Hb, rsum, rssq, g1, be1, H1D, B_ROWS * (H1D / 8));

  // layer 2: [B,4096] x [4096,2048]
  zero_f32<<<cdiv(B_ROWS, 256), 256, 0, stream>>>(rsum, B_ROWS);
  zero_f32<<<cdiv(B_ROWS, 256), 256, 0, stream>>>(rssq, B_ROWS);
  gemm_bf16_gelu_stats<<<dim3(B_ROWS / 64, H2D / 256), 256, 0, stream>>>(
      Hb, W2b, b2, Eb, rsum, rssq, H1D, H2D);
  ln_apply<<<cdiv((long)B_ROWS * H2D / 8, 256), 256, 0, stream>>>(
      Eb, rsum, rssq, g2, be2, H2D, B_ROWS * (H2D / 8));

  // layer 3: [B,2048] x [2048,1024]
  zero_f32<<<cdiv(B_ROWS, 256), 256, 0, stream>>>(rsum, B_ROWS);
  zero_f32<<<cdiv(B_ROWS, 256), 256, 0, stream>>>(rssq, B_ROWS);
  gemm_bf16_gelu_stats<<<dim3(B_ROWS / 64, H3D / 256), 256, 0, stream>>>(
      Eb, W3b, b3, H3b, rsum, rssq, H2D, H3D);
  ln_apply<<<cdiv((long)B_ROWS * H3D / 8, 256), 256, 0, stream>>>(
      H3b, rsum, rssq, g3, be3, H3D, B_ROWS * (H3D / 8));

  // head: logits + mask + softmax + argmax
  head_softmax<<<B_ROWS / 32, 256, 0, stream>>>(H3b, W4, b4, amask, pid, (float*)d_out);
}
